// TransformerConvBlock_19885698580756
// MI455X (gfx1250) — matrix operand order repaired and checked by reading, not compiled, not run on hardware
//
#include <hip/hip_runtime.h>
#include <cstdint>
#include <cstddef>

typedef __attribute__((ext_vector_type(16))) __bf16 v16bf;
typedef __attribute__((ext_vector_type(8)))  float  v8f;
typedef __attribute__((ext_vector_type(8)))  unsigned int v8u;

// ---------- helpers ----------
static __device__ __forceinline__ unsigned f2bf(float f) {
    unsigned u = __float_as_uint(f);
    u += 0x7fffu + ((u >> 16) & 1u);   // round-to-nearest-even
    return u >> 16;
}
static __device__ __forceinline__ unsigned pack_bf2(float lo, float hi) {
    return f2bf(lo) | (f2bf(hi) << 16);
}
// order-preserving float<->uint key (monotonic under unsigned compare)
static __device__ __forceinline__ unsigned fkey(float f) {
    unsigned b = __float_as_uint(f);
    return (b & 0x80000000u) ? ~b : (b | 0x80000000u);
}
static __device__ __forceinline__ float fdec(unsigned k) {
    unsigned b = (k & 0x80000000u) ? (k ^ 0x80000000u) : ~k;
    return __uint_as_float(b);
}

// ---------- generic bf16 WMMA GEMM:  C[M,NO] = act(A[M,KD] @ W[KD,NO] + bias) ----------
// LDS holds bf16 PAIRS packed along K in u32 words:
//   lB[n*SB2 + p] = {W[2p][n], W[2p+1][n]}   (transposed weight)
//   lA[r*SA2 + p] = {A[r][2p], A[r][2p+1]}
// so each lane's WMMA fragment is contiguous u32 runs -> ds_load_b128 merges.
template<int KD, int NO, int ROWS, bool LEAKY>
__global__ __launch_bounds__(256) void gemm_bf16_wmma(const float* __restrict__ A,
                                                      const float* __restrict__ W,
                                                      const float* __restrict__ bias,
                                                      float* __restrict__ C, int M)
{
    constexpr int SA2 = KD / 2 + 2;      // padded pair-stride (even -> 8B-aligned uint2 stores)
    constexpr int SB2 = KD / 2 + 2;
    __shared__ unsigned int lA[ROWS * SA2];
    __shared__ unsigned int lB[NO * SB2];
    const int tid = threadIdx.x;
    const int m0  = blockIdx.x * ROWS;

    // stage W transposed+packed: float4 loads (coalesced), 4 packed u32 stores
    constexpr int WITEMS = (KD / 2) * (NO / 4);
    for (int idx = tid; idx < WITEMS; idx += 256) {
        int p  = idx / (NO / 4);             // k-pair index
        int n4 = (idx - p * (NO / 4)) * 4;
        int k  = p * 2;
        float4 w0 = *(const float4*)(W + (size_t)k * NO + n4);
        float4 w1 = *(const float4*)(W + (size_t)(k + 1) * NO + n4);
        lB[(n4 + 0) * SB2 + p] = pack_bf2(w0.x, w1.x);
        lB[(n4 + 1) * SB2 + p] = pack_bf2(w0.y, w1.y);
        lB[(n4 + 2) * SB2 + p] = pack_bf2(w0.z, w1.z);
        lB[(n4 + 3) * SB2 + p] = pack_bf2(w0.w, w1.w);
    }
    // stage A rows: float4 loads, uint2 packed stores
    constexpr int AITEMS = ROWS * (KD / 4);
    for (int idx = tid; idx < AITEMS; idx += 256) {
        int r  = idx / (KD / 4);
        int k4 = (idx - r * (KD / 4)) * 4;
        int row = m0 + r;
        float4 a = make_float4(0.f, 0.f, 0.f, 0.f);
        if (row < M) a = *(const float4*)(A + (size_t)row * KD + k4);
        uint2 pk;
        pk.x = pack_bf2(a.x, a.y);
        pk.y = pack_bf2(a.z, a.w);
        *(uint2*)&lA[r * SA2 + k4 / 2] = pk;
    }
    __syncthreads();

    const int w = tid >> 5, l = tid & 31;
    const int half = l >> 4, l16 = l & 15;
    const int h4 = half * 4;
    constexpr int NTC = NO / 16;                 // tiles along N
    constexpr int NT  = (ROWS / 16) * NTC;       // total 16x16 tiles
    constexpr int TPW = NT / 8;                  // tiles per wave (8 waves)
    v8f acc[TPW];
    for (int t = 0; t < TPW; ++t)
        for (int i = 0; i < 8; ++i) acc[t][i] = 0.0f;

    for (int ks2 = 0; ks2 < KD / 2; ks2 += 16) {   // 32 K-elements per step (16 pairs)
        for (int t = 0; t < TPW; ++t) {
            const int tile = w + t * 8;
            const int rs = tile / NTC, nc = tile - rs * NTC;
            // A fragment: pairs [h4..h4+3] and [8+h4..8+h4+3]  (two b128 runs)
            const unsigned int* Ap = &lA[(rs * 16 + l16) * SA2 + ks2];
            v8u au;
#pragma unroll
            for (int j = 0; j < 4; ++j) au[j]     = Ap[h4 + j];
#pragma unroll
            for (int j = 0; j < 4; ++j) au[4 + j] = Ap[8 + h4 + j];
            // B fragment: pairs [h4..h4+3] and [8+h4..8+h4+3]  (two b128 runs)
            const unsigned int* Bp = &lB[(nc * 16 + l16) * SB2 + ks2 + h4];
            v8u bu;
#pragma unroll
            for (int j = 0; j < 8; ++j) bu[j] = Bp[(j & 3) + (j >> 2) * 8];
            acc[t] = __builtin_amdgcn_wmma_f32_16x16x32_bf16(
                false, __builtin_bit_cast(v16bf, au),
                false, __builtin_bit_cast(v16bf, bu),
                (short)0, acc[t], false, false);
        }
    }

    for (int t = 0; t < TPW; ++t) {
        const int tile = w + t * 8;
        const int rs = tile / NTC, nc = tile - rs * NTC;
        const int col = nc * 16 + l16;
        const float bv = bias ? bias[col] : 0.0f;
#pragma unroll
        for (int r = 0; r < 8; ++r) {
            int row = m0 + rs * 16 + half * 8 + r;   // C layout: VGPR r -> M = r + 8*half
            if (row < M) {
                float v = acc[t][r] + bv;
                if (LEAKY) v = (v > 0.0f) ? v : 0.01f * v;
                C[(size_t)row * NO + col] = v;
            }
        }
    }
}

// ---------- per-edge attention logits + segment max ----------
__global__ __launch_bounds__(256) void edge_logits_kernel(const int* __restrict__ ei,
        const float* __restrict__ Q, const float* __restrict__ Kp,
        const float* __restrict__ Ep, float* __restrict__ logits,
        unsigned* __restrict__ segmax, int E)
{
    int e = blockIdx.x * 8 + (threadIdx.x >> 5);
    if (e >= E) return;
    int l = threadIdx.x & 31;
    int src = ei[e], dst = ei[E + e];
    float4 q  = ((const float4*)(Q  + (size_t)dst * 128))[l];
    float4 k  = ((const float4*)(Kp + (size_t)src * 128))[l];
    float4 ee = ((const float4*)(Ep + (size_t)e   * 128))[l];
    float d = q.x*(k.x+ee.x) + q.y*(k.y+ee.y) + q.z*(k.z+ee.z) + q.w*(k.w+ee.w);
    // reduce within 8-lane head groups (head = lane>>3)
    d += __shfl_xor(d, 1, 32);
    d += __shfl_xor(d, 2, 32);
    d += __shfl_xor(d, 4, 32);
    if ((l & 7) == 0) {
        int h = l >> 3;
        float a = d * 0.17677669529663687f;   // 1/sqrt(32)
        logits[(size_t)e * 4 + h] = a;
        atomicMax(&segmax[(size_t)dst * 4 + h], fkey(a));
    }
}

// ---------- exp(logit - max) + segment sum ----------
__global__ __launch_bounds__(256) void edge_exp_kernel(const int* __restrict__ ei,
        const float* __restrict__ logits, const unsigned* __restrict__ segmax,
        float* __restrict__ expv, float* __restrict__ segsum, int E)
{
    size_t i = (size_t)blockIdx.x * 256 + threadIdx.x;
    if (i >= (size_t)E * 4) return;
    int e = (int)(i >> 2), h = (int)(i & 3);
    int dst = ei[E + e];
    float m = fdec(segmax[(size_t)dst * 4 + h]);
    float v = __expf(logits[i] - m);
    expv[i] = v;
    atomicAdd(&segsum[(size_t)dst * 4 + h], v);
}

// ---------- normalized message scatter-add ----------
__global__ __launch_bounds__(256) void scatter_msg_kernel(const int* __restrict__ ei,
        const float* __restrict__ V, const float* __restrict__ Ep,
        const float* __restrict__ expv, const float* __restrict__ segsum,
        float* __restrict__ agg, int E)
{
    int e = blockIdx.x * 8 + (threadIdx.x >> 5);
    if (e >= E) return;
    int l = threadIdx.x & 31;
    int src = ei[e], dst = ei[E + e];
    int h = l >> 3;
    float coeff = expv[(size_t)e * 4 + h] / (segsum[(size_t)dst * 4 + h] + 1e-16f);
    float4 v  = ((const float4*)(V  + (size_t)src * 128))[l];
    float4 ee = ((const float4*)(Ep + (size_t)e   * 128))[l];
    float* p = agg + (size_t)dst * 128 + l * 4;
    atomicAdd(p + 0, (v.x + ee.x) * coeff);
    atomicAdd(p + 1, (v.y + ee.y) * coeff);
    atomicAdd(p + 2, (v.z + ee.z) * coeff);
    atomicAdd(p + 3, (v.w + ee.w) * coeff);
}

// ---------- LayerNorm(a + b [+ c]) ----------
__global__ __launch_bounds__(256) void ln_kernel(const float* __restrict__ a,
        const float* __restrict__ b, const float* __restrict__ c,
        const float* __restrict__ g, const float* __restrict__ beta,
        float* __restrict__ out, int M)
{
    int row = blockIdx.x * 8 + (threadIdx.x >> 5);
    if (row >= M) return;
    int l = threadIdx.x & 31;
    float4 x = ((const float4*)(a + (size_t)row * 128))[l];
    float4 y = ((const float4*)(b + (size_t)row * 128))[l];
    x.x += y.x; x.y += y.y; x.z += y.z; x.w += y.w;
    if (c) {
        float4 z = ((const float4*)(c + (size_t)row * 128))[l];
        x.x += z.x; x.y += z.y; x.z += z.z; x.w += z.w;
    }
    float s1 = x.x + x.y + x.z + x.w;
    float s2 = x.x*x.x + x.y*x.y + x.z*x.z + x.w*x.w;
#pragma unroll
    for (int m = 1; m < 32; m <<= 1) {
        s1 += __shfl_xor(s1, m, 32);
        s2 += __shfl_xor(s2, m, 32);
    }
    float mu  = s1 * (1.0f / 128.0f);
    float var = s2 * (1.0f / 128.0f) - mu * mu;
    float rs  = rsqrtf(var + 1e-5f);
    float4 gg = ((const float4*)g)[l];
    float4 bb = ((const float4*)beta)[l];
    float4 o;
    o.x = (x.x - mu) * rs * gg.x + bb.x;
    o.y = (x.y - mu) * rs * gg.y + bb.y;
    o.z = (x.z - mu) * rs * gg.z + bb.z;
    o.w = (x.w - mu) * rs * gg.w + bb.w;
    ((float4*)(out + (size_t)row * 128))[l] = o;
}

__global__ void fill_u32x4_kernel(uint4* __restrict__ p, unsigned v, size_t n4) {
    size_t i = (size_t)blockIdx.x * 256 + threadIdx.x;
    if (i < n4) p[i] = make_uint4(v, v, v, v);
}

// ---------- host side ----------
extern "C" void kernel_launch(void* const* d_in, const int* in_sizes, int n_in,
                              void* d_out, int out_size, void* d_ws, size_t ws_size,
                              hipStream_t stream)
{
    const float* x     = (const float*)d_in[0];
    const int*   ei    = (const int*)  d_in[1];
    const float* eemb  = (const float*)d_in[2];
    const float* Wq    = (const float*)d_in[3];
    const float* bq    = (const float*)d_in[4];
    const float* Wk    = (const float*)d_in[5];
    const float* bk    = (const float*)d_in[6];
    const float* Wv    = (const float*)d_in[7];
    const float* bv    = (const float*)d_in[8];
    const float* We    = (const float*)d_in[9];
    const float* Wsk   = (const float*)d_in[10];
    const float* bsk   = (const float*)d_in[11];
    const float* g1    = (const float*)d_in[12];
    const float* be1   = (const float*)d_in[13];
    const float* W1    = (const float*)d_in[14];
    const float* b1    = (const float*)d_in[15];
    const float* W2    = (const float*)d_in[16];
    const float* b2    = (const float*)d_in[17];
    const float* g2    = (const float*)d_in[18];
    const float* be2   = (const float*)d_in[19];

    const int N = in_sizes[0] / 128;
    const int E = in_sizes[2] / 128;
    const size_t ND = (size_t)N * 128;

    float* ws      = (float*)d_ws;
    float* Qb      = ws;                    // [N,128]  (reused later as FFN2 output)
    float* Kb      = Qb + ND;               // [N,128]  (K..V region reused as FFN1 output [N,256])
    float* Vb      = Kb + ND;               // [N,128]
    float* Sb      = Vb + ND;               // [N,128] skip projection
    float* Ep      = Sb + ND;               // [E,128] edge projection
    float* logits  = Ep + (size_t)E * 128;  // [E,4]
    float* expv    = logits + (size_t)E * 4;// [E,4]
    float* segsum  = expv + (size_t)E * 4;  // [N,4]
    unsigned* segmax = (unsigned*)(segsum + (size_t)N * 4); // [N,4]
    float* agg     = (float*)segmax + (size_t)N * 4;        // [N,128]
    float* hb      = agg + ND;              // [N,128]
    float* t1      = Kb;                    // [N,256] reuse K+V
    float* fb      = Qb;                    // [N,128] reuse Q
    float* outp    = (float*)d_out;

    dim3 blk(256);
    // re-init accumulators every call (atomics mutate them)
    fill_u32x4_kernel<<<(unsigned)((ND / 4 + 255) / 256), blk, 0, stream>>>((uint4*)agg, 0u, ND / 4);
    fill_u32x4_kernel<<<(unsigned)(((size_t)N + 255) / 256), blk, 0, stream>>>((uint4*)segsum, 0u, (size_t)N);
    fill_u32x4_kernel<<<(unsigned)(((size_t)N + 255) / 256), blk, 0, stream>>>((uint4*)segmax, 0x00800000u /* key(-FLT_MAX) */, (size_t)N);

    const unsigned gN = (unsigned)((N + 63) / 64);
    const unsigned gE = (unsigned)((E + 63) / 64);

    gemm_bf16_wmma<128,128,64,false><<<gN, blk, 0, stream>>>(x,    Wq,  bq,      Qb, N);
    gemm_bf16_wmma<128,128,64,false><<<gN, blk, 0, stream>>>(x,    Wk,  bk,      Kb, N);
    gemm_bf16_wmma<128,128,64,false><<<gN, blk, 0, stream>>>(x,    Wv,  bv,      Vb, N);
    gemm_bf16_wmma<128,128,64,false><<<gN, blk, 0, stream>>>(x,    Wsk, bsk,     Sb, N);
    gemm_bf16_wmma<128,128,64,false><<<gE, blk, 0, stream>>>(eemb, We,  nullptr, Ep, E);

    edge_logits_kernel<<<(unsigned)((E + 7) / 8), blk, 0, stream>>>(ei, Qb, Kb, Ep, logits, segmax, E);
    edge_exp_kernel<<<(unsigned)(((size_t)E * 4 + 255) / 256), blk, 0, stream>>>(ei, logits, segmax, expv, segsum, E);
    scatter_msg_kernel<<<(unsigned)((E + 7) / 8), blk, 0, stream>>>(ei, Vb, Ep, expv, segsum, agg, E);

    ln_kernel<<<(unsigned)((N + 7) / 8), blk, 0, stream>>>(x, agg, Sb, g1, be1, hb, N);
    gemm_bf16_wmma<128,256,64,true ><<<gN, blk, 0, stream>>>(hb, W1, b1, t1, N);
    gemm_bf16_wmma<256,128,64,false><<<gN, blk, 0, stream>>>(t1, W2, b2, fb, N);
    ln_kernel<<<(unsigned)((N + 7) / 8), blk, 0, stream>>>(hb, fb, nullptr, g2, be2, outp, N);
}
